// Predictor_22222160789951
// MI455X (gfx1250) — matrix operand order repaired and checked by reading, not compiled, not run on hardware
//
#include <hip/hip_runtime.h>

typedef unsigned short u16;
typedef __attribute__((ext_vector_type(16))) __bf16 v16bf;
typedef __attribute__((ext_vector_type(8)))  float  v8f;
typedef __attribute__((ext_vector_type(4)))  unsigned int v4u;
typedef __attribute__((ext_vector_type(8)))  int v8i;
typedef __attribute__((ext_vector_type(4)))  int v4i;

#define HDIM 1024
#define BDIM 64
#define UDIM 512
#define G4   4096
#define NBLK 32     // persistent blocks (each owns 32 hidden cols)
#define HPB  32
#define LDP  132    // padded LDS row stride in floats

// ---------------- helpers ----------------

__device__ __forceinline__ u16 f2bf(float f) {
  unsigned x = __float_as_uint(f);
  unsigned r = x + 0x7FFFu + ((x >> 16) & 1u);   // round-to-nearest-even
  return (u16)(r >> 16);
}

union FragBF { v16bf v; uint4 q[2]; };

// A fragment (16x32 bf16): lane holds row m=lane%16; K values are
// 16*half+0..15 with half = lane/16  -> two 16B loads.
__device__ __forceinline__ v16bf load_a_frag(const u16* row_k, int half) {
  FragBF f;
  f.q[0] = *(const uint4*)(row_k + 16 * half);
  f.q[1] = *(const uint4*)(row_k + 16 * half + 8);
  return f.v;
}

// B fragment (32x16 bf16): lane holds col n=lane%16; K run = 16*half..+15,
// contiguous in [n][k] storage -> one 32B (two 16B) load. Works for both
// global and LDS pointers (ds_load_b128 when the pointer is __shared__).
__device__ __forceinline__ v16bf load_b_frag(const u16* col_k, int half) {
  FragBF f;
  f.q[0] = *(const uint4*)(col_k + 16 * half);
  f.q[1] = *(const uint4*)(col_k + 16 * half + 8);
  return f.v;
}

__device__ __forceinline__ float sigm(float x) { return 1.0f / (1.0f + __expf(-x)); }
__device__ __forceinline__ float tanh_fast(float x) {
  x = fminf(fmaxf(x, -15.0f), 15.0f);
  float e = __expf(2.0f * x);
  return (e - 1.0f) / (e + 1.0f);
}

// ---------------- prep kernels ----------------

__global__ void k_f32_to_bf16(const float* __restrict__ src, u16* __restrict__ dst, int n) {
  int i = blockIdx.x * blockDim.x + threadIdx.x;
  if (i < n) dst[i] = f2bf(src[i]);
}

// W_comb[g][d] = sum_k W_ih[g][k] * W_embed[k][d];  b_comb = W_ih@b_embed + b_ih + b_hh
__global__ void k_fold_embed(const float* __restrict__ W_ih, const float* __restrict__ W_embed,
                             const float* __restrict__ b_embed, const float* __restrict__ b_ih,
                             const float* __restrict__ b_hh,
                             float* __restrict__ W_comb, float* __restrict__ b_comb) {
  int g = blockIdx.x * blockDim.x + threadIdx.x;
  if (g >= G4) return;
  const float* wr = W_ih + (size_t)g * HDIM;
  float a0 = 0.f, a1 = 0.f, a2 = 0.f, ab = 0.f;
  for (int k = 0; k < HDIM; ++k) {
    float w = wr[k];
    a0 += w * W_embed[k * 3 + 0];
    a1 += w * W_embed[k * 3 + 1];
    a2 += w * W_embed[k * 3 + 2];
    ab += w * b_embed[k];
  }
  W_comb[g * 3 + 0] = a0;
  W_comb[g * 3 + 1] = a1;
  W_comb[g * 3 + 2] = a2;
  b_comb[g] = ab + b_ih[g] + b_hh[g];
}

__global__ void k_init_state(const float* __restrict__ h0, const float* __restrict__ c0,
                             u16* __restrict__ h_buf, float* __restrict__ c_state,
                             unsigned* __restrict__ bar) {
  int i = blockIdx.x * blockDim.x + threadIdx.x;
  if (i < BDIM * HDIM) {
    int k = i & (HDIM - 1);
    h_buf[i] = f2bf(h0[k]);
    c_state[i] = c0[k];
  }
  if (i == 0) bar[0] = 0u;
}

// ---------------- persistent LSTM + fused output projection ----------------

__global__ void __launch_bounds__(256)
k_lstm_persistent(const float* __restrict__ y,        // [B][U][3]
                  const float* __restrict__ W_comb,   // [4096][3]
                  const float* __restrict__ b_comb,   // [4096]
                  const u16*  __restrict__ Whh,       // [4096][1024] bf16
                  const u16*  __restrict__ Wout,      // [1024][1024] bf16
                  const float* __restrict__ b_out,    // [1024]
                  u16*  __restrict__ h_buf,           // 2 x [B][H] bf16 (ping-pong)
                  float* __restrict__ c_state,        // [B][H]
                  float* __restrict__ out,            // [B][U][1024]
                  unsigned* __restrict__ bar) {
  // LDS budget (WGP has 320KB): 256KB weights + 33KB gates + ~2KB misc
  __shared__ __align__(16) u16 lds_w[128 * HDIM];   // this block's W_hh slice (bf16)
  __shared__ float gates_lds[BDIM * LDP];           // [64][128] gate pre-activations
  __shared__ float Wc_lds[128 * 3];
  __shared__ float bc_lds[128];

  const int t     = threadIdx.x;
  const int wave  = t >> 5;
  const int lane  = t & 31;
  const int half  = lane >> 4;
  const int l16   = lane & 15;
  const int hbase = blockIdx.x * HPB;       // this block's hidden-column base

  // ---- one-time: stage W_hh slice (4 gates x 32 rows x 1024 k, bf16) in LDS ----
  // Strided 3D tile: x=1024 cols, y=32 rows (stride 1024), z=4 gates (stride 1024*1024),
  // landing dense in LDS so row nl = gate*32 + hl matches B-fragment addressing.
  if (wave == 0) {
    unsigned long long ga = (unsigned long long)(const void*)(Whh + (size_t)hbase * HDIM);
    unsigned lds_off = (unsigned)(unsigned long long)(void*)&lds_w[0];
    // D# group 0: count=1, lds_addr, 57b global_addr, type=2
    v4u g0 = { 1u, lds_off, (unsigned)(ga & 0xffffffffu),
               (unsigned)((ga >> 32) & 0x01ffffffu) | 0x80000000u /* type=2 */ };
    // D# group 1
    v8i g1 = { (int)0x00010000u,              // workgroup_mask=0, data_size=1 (2B)
               (int)(1024u << 16),            // tensor_dim0[15:0] in [31:16]
               (int)(4096u << 16),            // tensor_dim0 hi=0, tensor_dim1 lo=4096
               (int)(1024u << 16),            // tensor_dim1 hi=0, tile_dim0=1024
               (int)((4u << 16) | 32u),       // tile_dim1=32, tile_dim2=4
               (int)1024u,                    // tensor_dim0_stride lo
               (int)0,                        // dim0_stride hi=0, dim1_stride lo16=0
               (int)16u };                    // tensor_dim1_stride[47:16] = 1048576>>16
    v4i g2 = { 4, 0, 0, 0 };                  // tensor_dim2=4; dim3/tile_dim3 unused
    v4i g3 = { 0, 0, 0, 0 };
    v8i gx = { 0, 0, 0, 0, 0, 0, 0, 0 };      // extension group (unused)
    __builtin_amdgcn_tensor_load_to_lds(g0, g1, g2, g3, gx, 0);
    __builtin_amdgcn_s_wait_tensorcnt(0);
  }

  // cache folded-embed weights for this block's 128 gate columns
  if (t < 128) {
    int g = t >> 5, hl = t & 31;
    int gcol = g * HDIM + hbase + hl;
    Wc_lds[t * 3 + 0] = W_comb[gcol * 3 + 0];
    Wc_lds[t * 3 + 1] = W_comb[gcol * 3 + 1];
    Wc_lds[t * 3 + 2] = W_comb[gcol * 3 + 2];
    bc_lds[t] = b_comb[gcol];
  }
  __syncthreads();

  for (int u = 0; u < UDIM; ++u) {
    const u16* hprev = h_buf + (size_t)(u & 1) * (BDIM * HDIM);
    u16*       hnext = h_buf + (size_t)((u + 1) & 1) * (BDIM * HDIM);

    // ---- phase 1: gates = h_prev @ Whh_sliceT  (bf16 WMMA, f32 acc, B from LDS) ----
    {
      const int nl = wave * 16 + l16;                 // local gate col 0..127
      const u16* wrow = lds_w + (size_t)nl * HDIM;    // LDS-resident weights
      const u16* a0p = hprev + (size_t)(0 * 16 + l16) * HDIM;
      const u16* a1p = hprev + (size_t)(1 * 16 + l16) * HDIM;
      const u16* a2p = hprev + (size_t)(2 * 16 + l16) * HDIM;
      const u16* a3p = hprev + (size_t)(3 * 16 + l16) * HDIM;
      v8f acc0 = {}, acc1 = {}, acc2 = {}, acc3 = {};
      for (int kb = 0; kb < HDIM; kb += 32) {
        v16bf bf = load_b_frag(wrow + kb, half);
        v16bf a0 = load_a_frag(a0p + kb, half);
        v16bf a1 = load_a_frag(a1p + kb, half);
        v16bf a2 = load_a_frag(a2p + kb, half);
        v16bf a3 = load_a_frag(a3p + kb, half);
        acc0 = __builtin_amdgcn_wmma_f32_16x16x32_bf16(false, a0, false, bf, (short)0, acc0, false, false);
        acc1 = __builtin_amdgcn_wmma_f32_16x16x32_bf16(false, a1, false, bf, (short)0, acc1, false, false);
        acc2 = __builtin_amdgcn_wmma_f32_16x16x32_bf16(false, a2, false, bf, (short)0, acc2, false, false);
        acc3 = __builtin_amdgcn_wmma_f32_16x16x32_bf16(false, a3, false, bf, (short)0, acc3, false, false);
      }
#pragma unroll
      for (int r = 0; r < 8; ++r) {
        gates_lds[(0 * 16 + r + 8 * half) * LDP + nl] = acc0[r];
        gates_lds[(1 * 16 + r + 8 * half) * LDP + nl] = acc1[r];
        gates_lds[(2 * 16 + r + 8 * half) * LDP + nl] = acc2[r];
        gates_lds[(3 * 16 + r + 8 * half) * LDP + nl] = acc3[r];
      }
    }
    __syncthreads();

    // ---- elementwise LSTM cell (64 batch x 32 hidden per block) ----
    {
      const int b  = t >> 2;
      const int h8 = (t & 3) * 8;
      const int upv = (u + UDIM - 1) & (UDIM - 1);   // x_seq = roll(y, 1)
      const float* xp = y + ((size_t)b * UDIM + upv) * 3;
      const float x0 = xp[0], x1 = xp[1], x2 = xp[2];
#pragma unroll
      for (int i = 0; i < 8; ++i) {
        const int hl = h8 + i;
        float gv[4];
#pragma unroll
        for (int g = 0; g < 4; ++g) {
          const int nl = g * 32 + hl;
          float gx = x0 * Wc_lds[nl * 3 + 0] + x1 * Wc_lds[nl * 3 + 1] +
                     x2 * Wc_lds[nl * 3 + 2] + bc_lds[nl];
          gv[g] = gates_lds[b * LDP + nl] + gx;
        }
        const float ig = sigm(gv[0]);
        const float fg = sigm(gv[1]);
        const float gg = tanh_fast(gv[2]);
        const float og = sigm(gv[3]);
        const int ci = b * HDIM + hbase + hl;
        const float cn = fg * c_state[ci] + ig * gg;
        c_state[ci] = cn;
        hnext[ci] = f2bf(og * tanh_fast(cn));
      }
    }

    // ---- device-wide barrier: everyone's h slice written before use ----
    __syncthreads();
    __threadfence();
    if (t == 0) {
      __hip_atomic_fetch_add(bar, 1u, __ATOMIC_RELEASE, __HIP_MEMORY_SCOPE_AGENT);
      const unsigned goal = (unsigned)NBLK * (unsigned)(u + 1);
      while (__hip_atomic_load(bar, __ATOMIC_ACQUIRE, __HIP_MEMORY_SCOPE_AGENT) < goal) {
        __builtin_amdgcn_s_sleep(2);
      }
    }
    __syncthreads();

    // ---- phase 2: out[:, u, cols] = h_u @ WoutT + b_out (fused projection) ----
    {
      const int mt  = wave & 3;
      const int nt  = wave >> 2;
      const int col = hbase + nt * 16 + l16;
      const u16* wrow = Wout + (size_t)col * HDIM;
      const u16* ap   = hnext + (size_t)(mt * 16 + l16) * HDIM;
      v8f acc = {};
      for (int kb = 0; kb < HDIM; kb += 32) {
        v16bf bf = load_b_frag(wrow + kb, half);
        v16bf af = load_a_frag(ap + kb, half);
        acc = __builtin_amdgcn_wmma_f32_16x16x32_bf16(false, af, false, bf, (short)0, acc, false, false);
      }
      const float bo = b_out[col];
#pragma unroll
      for (int r = 0; r < 8; ++r) {
        const int m = mt * 16 + r + 8 * half;  // batch index
        out[((size_t)m * UDIM + u) * HDIM + col] = acc[r] + bo;
      }
    }
  }
}

// ---------------- host ----------------

extern "C" void kernel_launch(void* const* d_in, const int* in_sizes, int n_in,
                              void* d_out, int out_size, void* d_ws, size_t ws_size,
                              hipStream_t stream) {
  const float* y       = (const float*)d_in[0];
  const float* W_embed = (const float*)d_in[1];
  const float* b_embed = (const float*)d_in[2];
  const float* W_ih    = (const float*)d_in[3];
  const float* W_hh    = (const float*)d_in[4];
  const float* b_ih    = (const float*)d_in[5];
  const float* b_hh    = (const float*)d_in[6];
  const float* W_out   = (const float*)d_in[7];
  const float* b_out   = (const float*)d_in[8];
  const float* h0      = (const float*)d_in[9];
  const float* c0      = (const float*)d_in[10];
  float* out = (float*)d_out;

  char* ws = (char*)d_ws;
  u16*      Whh_bf  = (u16*)(ws + 0);            // 4096*1024*2  = 8,388,608 B
  u16*      Wout_bf = (u16*)(ws + 8388608);      // 1024*1024*2  = 2,097,152 B
  float*    W_comb  = (float*)(ws + 10485760);   // 4096*3*4     = 49,152 B
  float*    b_comb  = (float*)(ws + 10534912);   // 4096*4       = 16,384 B
  u16*      h_buf   = (u16*)(ws + 10551296);     // 2*64*1024*2  = 262,144 B
  float*    c_state = (float*)(ws + 10813440);   // 64*1024*4    = 262,144 B
  unsigned* bar     = (unsigned*)(ws + 11075584);

  k_f32_to_bf16<<<(G4 * HDIM + 255) / 256, 256, 0, stream>>>(W_hh, Whh_bf, G4 * HDIM);
  k_f32_to_bf16<<<(HDIM * HDIM + 255) / 256, 256, 0, stream>>>(W_out, Wout_bf, HDIM * HDIM);
  k_fold_embed<<<G4 / 256, 256, 0, stream>>>(W_ih, W_embed, b_embed, b_ih, b_hh, W_comb, b_comb);
  k_init_state<<<(BDIM * HDIM + 255) / 256, 256, 0, stream>>>(h0, c0, h_buf, c_state, bar);

  k_lstm_persistent<<<NBLK, 256, 0, stream>>>(y, W_comb, b_comb, Whh_bf, Wout_bf, b_out,
                                              h_buf, c_state, out, bar);
}
